// Transformer_70171175682594
// MI455X (gfx1250) — matrix operand order repaired and checked by reading, not compiled, not run on hardware
//
#include <hip/hip_runtime.h>
#include <hip/hip_bf16.h>
#include <stdint.h>

// ---------------------------------------------------------------------------
// Transformer forward (enc-dec) for gfx1250.
// GEMMs run on v_wmma_f32_16x16x32_bf16 (bf16 inputs, f32 accumulate).
// Software-pipelined: double-buffered LDS tiles, register-staged global loads,
// weight tiles streamed via global_load_async_to_lds_b32 (ASYNCcnt).
// All kernels stream-ordered; no sync/alloc APIs (graph-capture safe).
// ---------------------------------------------------------------------------

#define USE_ASYNC_LDS 1

typedef __attribute__((ext_vector_type(16))) __bf16 v16bf;
typedef __attribute__((ext_vector_type(8)))  float  v8f;

#define TB   8            // batch
#define SEQ  512          // LS == LT
#define DM   512          // model dim
#define NH   8            // heads
#define HD   64           // head dim
#define DFF  2048
#define NL   6
#define NV   32000

#define BM 128
#define BK 32
#define A_STRIDE 17       // BK/2 + 1 dwords per row (bank-conflict pad)

// B-source kinds for the GEMM template
#define B_F32N 0          // f32 row-major [K][N]
#define B_F32T 1          // f32 transposed: B[k][n] = src[n*ldb + k]
#define B_PKN  2          // bf16-pair dwords [K/2][N] (ldb in dwords)
#define B_PKT  3          // bf16-pair dwords [N][K/2] (ldb in dwords)

// ---------------- input flattening (setup_inputs dict order) ----------------
enum {
  IN_SRC = 0, IN_TRG = 1, IN_EMB = 2,
  EA_WQ = 3, EA_WK, EA_WV, EA_WO, EA_G, EA_B,          // enc_attn
  EF_W1 = 9, EF_B1, EF_W2, EF_B2, EF_G, EF_B,          // enc_ffn
  IN_ENC_G = 15, IN_ENC_B = 16,
  DS_WQ = 17, DS_WK, DS_WV, DS_WO, DS_G, DS_B,         // dec_sattn
  DX_WQ = 23, DX_WK, DX_WV, DX_WO, DX_G, DX_B,         // dec_xattn
  DF_W1 = 29, DF_B1, DF_W2, DF_B2, DF_G, DF_B,         // dec_ffn
  IN_DEC_G = 35, IN_DEC_B = 36
};

// ---------------------------------------------------------------------------
__device__ __forceinline__ uint32_t pack_bf16(float a, float b) {
  uint32_t x = __float_as_uint(a), y = __float_as_uint(b);
  x += 0x7FFFu + ((x >> 16) & 1u);                       // RNE
  y += 0x7FFFu + ((y >> 16) & 1u);
  // take high halves of x (low result) and y (high result): one v_perm_b32
  return __builtin_amdgcn_perm(y, x, 0x07060302u);
}

#if USE_ASYNC_LDS
__device__ __forceinline__ void async_ld_b32(uint32_t lds_off,
                                             const uint32_t* gsrc) {
  // global -> LDS async copy, tracked by ASYNCcnt (CDNA5).
  asm volatile("global_load_async_to_lds_b32 %0, %1, off"
               :: "v"(lds_off), "v"(gsrc) : "memory");
}
template <int N>
__device__ __forceinline__ void wait_async_le() {
  asm volatile("s_wait_asynccnt %0" :: "i"(N) : "memory");
}
// flat shared-pointer truncates to the AS(3) byte offset (aperture low32 == 0)
__device__ __forceinline__ uint32_t lds_off32(const void* p) {
  return (uint32_t)(uintptr_t)p;
}
#endif

// C = alpha * A(MxK) * B(KxN) [+ bias] [+ Res] [ReLU]; bf16 WMMA, f32 accum.
// 2-level batching: z -> (outer = z / batInner, inner = z % batInner).
// Block tile: BM(128) x BNT, 8 waves (4 along M x 2 along N),
// wave tile 32 x (BNT/2) = 2 x (BNT/32) sub-tiles of 16x16.
// Pipeline: double-buffered LDS; next tile's global loads / async copies are
// issued before the barrier so they are in flight during the WMMA phase.
template <int BNT, int BSRC, bool HASBIAS, bool HASRES, bool RELU>
__global__ __launch_bounds__(256) void gemm_wmma_bf16(
    const float* __restrict__ A, const void* __restrict__ Bv,
    float* __restrict__ C, const float* __restrict__ bias,
    const float* __restrict__ Res,
    int K, int lda, int ldb, int ldc, int batInner,
    long long aO, long long aI, long long bO, long long bI,
    long long cO, long long cI, float alpha)
{
  constexpr int SN = BNT / 32;          // N sub-tiles per wave
  constexpr int B_STRIDE = BNT + 1;     // dwords per kh row (bank pad)
  constexpr int NBI = BNT / 16;         // B staging iterations / thread

  __shared__ uint32_t sA[2 * BM * A_STRIDE];        // ping-pong A tiles
  __shared__ uint32_t sB[2 * (BK / 2) * B_STRIDE];  // ping-pong B tiles

  const int tid = threadIdx.x;
  const int z   = blockIdx.z;
  const int zo  = z / batInner, zi = z % batInner;
  A += zo * aO + zi * aI;
  const float*    Bf = (const float*)Bv;
  const uint32_t* Bp = (const uint32_t*)Bv;
  if constexpr (BSRC <= 1) Bf += zo * bO + zi * bI;
  else                     Bp += zo * bO + zi * bI;
  const long long coff = zo * cO + zi * cI;
  C += coff;
  if (HASRES) Res += coff;

  const int m0 = blockIdx.y * BM;
  const int n0 = blockIdx.x * BNT;

  const int lane = tid & 31;
  const int wave = tid >> 5;
  const int wm = wave >> 1;       // 4 waves along M
  const int wn = wave & 1;        // 2 waves along N
  const int hlf = lane >> 4;      // lane half (0/1)
  const int lr  = lane & 15;

  v8f acc[2][SN];
  #pragma unroll
  for (int i = 0; i < 2; ++i)
    #pragma unroll
    for (int j = 0; j < SN; ++j)
      #pragma unroll
      for (int r = 0; r < 8; ++r) acc[i][j][r] = 0.f;

  // ---- register staging buffers ----
  float2   avr[8];                 // A tile f32 pairs
  float2   btr[NBI];               // B f32 transposed pairs
  float    b0r[NBI], b1r[NBI];     // B f32 row-major pair halves
  uint32_t bpr[NBI];               // B packed dwords (non-async fallback)

  auto load_A = [&](int k0) {
    #pragma unroll
    for (int i = 0; i < 8; ++i) {
      const int p = tid + i * 256;
      avr[i] = *reinterpret_cast<const float2*>(
          A + (size_t)(m0 + (p >> 4)) * lda + k0 + 2 * (p & 15));
    }
  };
  auto store_A = [&](uint32_t* d) {
    #pragma unroll
    for (int i = 0; i < 8; ++i) {
      const int p = tid + i * 256;
      d[(p >> 4) * A_STRIDE + (p & 15)] = pack_bf16(avr[i].x, avr[i].y);
    }
  };
  auto load_B = [&](int k0) {
    #pragma unroll
    for (int i = 0; i < NBI; ++i) {
      const int p = tid + i * 256;
      const int n = p % BNT, kh = p / BNT;
      if constexpr (BSRC == B_F32N) {
        const float* s = Bf + (size_t)(k0 + 2 * kh) * ldb + n0 + n;
        b0r[i] = s[0]; b1r[i] = s[ldb];
      } else if constexpr (BSRC == B_F32T) {
        btr[i] = *reinterpret_cast<const float2*>(
            Bf + (size_t)(n0 + n) * ldb + k0 + 2 * kh);
      }
#if !USE_ASYNC_LDS
      else if constexpr (BSRC == B_PKN) {
        bpr[i] = Bp[(size_t)((k0 >> 1) + kh) * ldb + n0 + n];
      } else {
        bpr[i] = Bp[(size_t)(n0 + n) * ldb + (k0 >> 1) + kh];
      }
#endif
    }
  };
  auto store_B = [&](uint32_t* d) {
    #pragma unroll
    for (int i = 0; i < NBI; ++i) {
      const int p = tid + i * 256;
      const int n = p % BNT, kh = p / BNT;
      if constexpr (BSRC == B_F32N)      d[kh * B_STRIDE + n] = pack_bf16(b0r[i], b1r[i]);
      else if constexpr (BSRC == B_F32T) d[kh * B_STRIDE + n] = pack_bf16(btr[i].x, btr[i].y);
#if !USE_ASYNC_LDS
      else                               d[kh * B_STRIDE + n] = bpr[i];
#endif
    }
  };
#if USE_ASYNC_LDS
  auto async_B = [&](int k0, uint32_t* d) {
    if constexpr (BSRC >= B_PKN) {
      #pragma unroll
      for (int i = 0; i < NBI; ++i) {
        const int p = tid + i * 256;
        const int n = p % BNT, kh = p / BNT;
        const uint32_t* s = (BSRC == B_PKN)
            ? Bp + (size_t)((k0 >> 1) + kh) * ldb + n0 + n
            : Bp + (size_t)(n0 + n) * ldb + (k0 >> 1) + kh;
        async_ld_b32(lds_off32(&d[kh * B_STRIDE + n]), s);
      }
    }
  };
#endif

  // ---- prologue: tile 0 in flight ----
  load_A(0);
  load_B(0);
#if USE_ASYNC_LDS
  async_B(0, sB);
#endif

  const int ksteps = K / BK;
  for (int kt = 0; kt < ksteps; ++kt) {
    const int cur = kt & 1;
    uint32_t* sAc = sA + cur * (BM * A_STRIDE);
    uint32_t* sBc = sB + cur * ((BK / 2) * B_STRIDE);
    uint32_t* sBn = sB + (cur ^ 1) * ((BK / 2) * B_STRIDE);

    // publish current tile (registers were loaded an iteration ago)
    store_A(sAc);
    if constexpr (BSRC <= B_F32T) store_B(sBc);
#if !USE_ASYNC_LDS
    else store_B(sBc);
#endif

    // launch next tile's global traffic before the barrier
    const bool more = (kt + 1) < ksteps;
    if (more) {
      load_A(kt * BK + BK);
      load_B(kt * BK + BK);
#if USE_ASYNC_LDS
      async_B(kt * BK + BK, sBn);
#endif
    }
#if USE_ASYNC_LDS
    if constexpr (BSRC >= B_PKN) {
      // wait only for the older tile's copies; keep next tile's in flight
      if (more) wait_async_le<NBI>(); else wait_async_le<0>();
    }
#endif
    __syncthreads();

    // ---- gather fragments per ISA layout ----
    union Frag { uint32_t u[8]; v16bf v; };
    Frag fa[2], fb[SN];
    #pragma unroll
    for (int sm = 0; sm < 2; ++sm) {
      const int m = wm * 32 + sm * 16 + lr;
      #pragma unroll
      for (int j = 0; j < 8; ++j) {
        // A 16x32 bf16: vgpr j holds K = (j<4?0:16) + half*8 + (j&3)*2 .. +1
        const int kk2 = (j < 4 ? 0 : 8) + hlf * 4 + (j & 3);
        fa[sm].u[j] = sAc[m * A_STRIDE + kk2];
      }
    }
    #pragma unroll
    for (int sn = 0; sn < SN; ++sn) {
      const int n = wn * (BNT / 2) + sn * 16 + lr;
      #pragma unroll
      for (int j = 0; j < 8; ++j) {
        // B 32x16 bf16: vgpr j holds K = (j<4?0:16) + half*8 + (j&3)*2 .. +1 at column n
        fb[sn].u[j] = sBc[((j < 4 ? 0 : 8) + hlf * 4 + (j & 3)) * B_STRIDE + n];
      }
    }
    #pragma unroll
    for (int sm = 0; sm < 2; ++sm)
      #pragma unroll
      for (int sn = 0; sn < SN; ++sn)
        acc[sm][sn] = __builtin_amdgcn_wmma_f32_16x16x32_bf16(
            false, fa[sm].v, false, fb[sn].v, (short)0, acc[sm][sn],
            false, false);
    // no trailing barrier: next iteration writes the other LDS buffer
  }

  // ---- epilogue: alpha, bias, residual, relu (all compile-time flags) ----
  #pragma unroll
  for (int sm = 0; sm < 2; ++sm) {
    const int rowb = m0 + wm * 32 + sm * 16 + hlf * 8;
    #pragma unroll
    for (int sn = 0; sn < SN; ++sn) {
      const int col = n0 + wn * (BNT / 2) + sn * 16 + lr;
      float bval = 0.f;
      if constexpr (HASBIAS) bval = bias[col];
      float*       cp = C + (size_t)rowb * ldc + col;
      const float* rp = HASRES ? (Res + (size_t)rowb * ldc + col) : nullptr;
      #pragma unroll
      for (int r = 0; r < 8; ++r) {
        float v = acc[sm][sn][r] * alpha;
        if constexpr (HASBIAS) v += bval;
        if constexpr (HASRES)  v += rp[(size_t)r * ldc];
        if constexpr (RELU)    v = v > 0.f ? v : 0.f;
        cp[(size_t)r * ldc] = v;
      }
    }
  }
}

// ---- weight pre-pack: f32 [rows][N] -> bf16-pair dwords [rows/2][N] --------
__global__ __launch_bounds__(256) void pack_rows_kernel(
    uint32_t* __restrict__ out, const float* __restrict__ in,
    int N, long long pairs)
{
  const long long p = (long long)blockIdx.x * 256 + threadIdx.x;
  if (p >= pairs) return;
  const long long kh = p / N;
  const int       n  = (int)(p % N);
  const float* s = in + (size_t)(2 * kh) * N + n;
  out[p] = pack_bf16(s[0], s[N]);
}

// ---- flat pre-pack: f32 [2*pairs] -> bf16-pair dwords (adjacent elements) --
__global__ __launch_bounds__(256) void pack_flat_kernel(
    uint32_t* __restrict__ out, const float* __restrict__ in, long long pairs)
{
  const long long p = (long long)blockIdx.x * 256 + threadIdx.x;
  if (p >= pairs) return;
  const float2 v = *reinterpret_cast<const float2*>(in + 2 * p);
  out[p] = pack_bf16(v.x, v.y);
}

// ---- masked softmax over rows of length 512; one wave per row --------------
__global__ __launch_bounds__(256) void softmax_mask_kernel(
    float* __restrict__ S, const int* __restrict__ seq, int causal)
{
  const int row  = blockIdx.x * 8 + (threadIdx.x >> 5); // row in [0, B*H*SEQ)
  const int lane = threadIdx.x & 31;
  const int q  = row % SEQ;
  const int bh = row / SEQ;
  const int b  = bh / NH;
  float* p = S + (size_t)row * SEQ;
  const int* ks = seq + (size_t)b * SEQ;

  float v[16];
  float mx = -3.0e38f;
  #pragma unroll
  for (int i = 0; i < 16; ++i) {
    const int c = lane + i * 32;
    float s = p[c];
    const bool ok = (ks[c] != 0) && (!causal || c <= q);
    s = ok ? s : -1e9f;
    v[i] = s;
    mx = fmaxf(mx, s);
  }
  #pragma unroll
  for (int o = 16; o > 0; o >>= 1) mx = fmaxf(mx, __shfl_xor(mx, o, 32));
  float sum = 0.f;
  #pragma unroll
  for (int i = 0; i < 16; ++i) { v[i] = __expf(v[i] - mx); sum += v[i]; }
  #pragma unroll
  for (int o = 16; o > 0; o >>= 1) sum += __shfl_xor(sum, o, 32);
  const float inv = 1.f / sum;
  #pragma unroll
  for (int i = 0; i < 16; ++i) p[lane + i * 32] = v[i] * inv;
}

// ---- LayerNorm over rows of 512; one wave per row; in-place safe -----------
__global__ __launch_bounds__(256) void layernorm_kernel(
    float* __restrict__ X, const float* __restrict__ g,
    const float* __restrict__ b)
{
  const int row  = blockIdx.x * 8 + (threadIdx.x >> 5);
  const int lane = threadIdx.x & 31;
  float* p = X + (size_t)row * DM;
  float v[16];
  float s = 0.f;
  #pragma unroll
  for (int i = 0; i < 16; ++i) { v[i] = p[lane + i * 32]; s += v[i]; }
  #pragma unroll
  for (int o = 16; o > 0; o >>= 1) s += __shfl_xor(s, o, 32);
  const float mu = s * (1.f / DM);
  float ss = 0.f;
  #pragma unroll
  for (int i = 0; i < 16; ++i) { const float d = v[i] - mu; ss += d * d; }
  #pragma unroll
  for (int o = 16; o > 0; o >>= 1) ss += __shfl_xor(ss, o, 32);
  const float inv = rsqrtf(ss * (1.f / DM) + 1e-6f);
  #pragma unroll
  for (int i = 0; i < 16; ++i) {
    const int c = lane + i * 32;
    p[c] = (v[i] - mu) * inv * g[c] + b[c];
  }
}

// ---- embedding gather + sinusoidal positional encoding ---------------------
__global__ __launch_bounds__(256) void embed_pos_kernel(
    float* __restrict__ X, const float* __restrict__ emb,
    const int* __restrict__ seq)
{
  const size_t idx = (size_t)blockIdx.x * 256 + threadIdx.x; // B*SEQ*DM total
  const int    d   = (int)(idx & (DM - 1));
  const size_t row = idx >> 9;
  const int    l   = (int)(row % SEQ);
  const int    tok = seq[row];
  const double e   = (double)(2 * (d / 2)) / (double)DM;
  const double ang = (double)l / pow(10000.0, e);
  const float  pe  = (d & 1) ? (float)cos(ang) : (float)sin(ang);
  X[idx] = emb[(size_t)tok * DM + d] + pe;
}

// ---------------------------------------------------------------------------
// host-side helpers
// ---------------------------------------------------------------------------
static void launch_gemm(hipStream_t st, const float* A, const void* B,
                        float* C, const float* bias, const float* res,
                        int M, int N, int K, int lda, int ldb, int ldc,
                        int nbat, int batInner,
                        long long aO, long long aI, long long bO, long long bI,
                        long long cO, long long cI,
                        float alpha, int bsrc, bool relu)
{
  const int bnt = (N % 128 == 0) ? 128 : 64;
  dim3 grid(N / bnt, M / BM, nbat), blk(256);
  const bool hb = bias != nullptr, hr = res != nullptr;
  if (bnt == 64) {                               // P.V context (f32 B)
    gemm_wmma_bf16<64, B_F32N, false, false, false><<<grid, blk, 0, st>>>(
        A, B, C, bias, res, K, lda, ldb, ldc, batInner,
        aO, aI, bO, bI, cO, cI, alpha);
  } else if (bsrc == B_F32T) {                   // attention scores
    gemm_wmma_bf16<128, B_F32T, false, false, false><<<grid, blk, 0, st>>>(
        A, B, C, bias, res, K, lda, ldb, ldc, batInner,
        aO, aI, bO, bI, cO, cI, alpha);
  } else if (bsrc == B_PKT) {                    // final y @ emb^T
    gemm_wmma_bf16<128, B_PKT, false, false, false><<<grid, blk, 0, st>>>(
        A, B, C, bias, res, K, lda, ldb, ldc, batInner,
        aO, aI, bO, bI, cO, cI, alpha);
  } else if (hb && relu) {                       // FFN1
    gemm_wmma_bf16<128, B_PKN, true, false, true><<<grid, blk, 0, st>>>(
        A, B, C, bias, res, K, lda, ldb, ldc, batInner,
        aO, aI, bO, bI, cO, cI, alpha);
  } else if (hb && hr) {                         // FFN2
    gemm_wmma_bf16<128, B_PKN, true, true, false><<<grid, blk, 0, st>>>(
        A, B, C, bias, res, K, lda, ldb, ldc, batInner,
        aO, aI, bO, bI, cO, cI, alpha);
  } else if (hr) {                               // Wo + residual
    gemm_wmma_bf16<128, B_PKN, false, true, false><<<grid, blk, 0, st>>>(
        A, B, C, bias, res, K, lda, ldb, ldc, batInner,
        aO, aI, bO, bI, cO, cI, alpha);
  } else {                                       // QKV projections
    gemm_wmma_bf16<128, B_PKN, false, false, false><<<grid, blk, 0, st>>>(
        A, B, C, bias, res, K, lda, ldb, ldc, batInner,
        aO, aI, bO, bI, cO, cI, alpha);
  }
}

struct Work {
  float *QH, *KH, *VH, *AO, *X, *Y, *ATT, *HID;
  uint32_t *pkAttn[12];   // ea q,k,v,o | ds q,k,v,o | dx q,k,v,o (all layers)
  uint32_t *pkW1e, *pkW2e, *pkW1d, *pkW2d, *pkEmb;
};

static void run_mha(hipStream_t st, float* x, const float* kvin,
                    const uint32_t* wq, const uint32_t* wk,
                    const uint32_t* wv, const uint32_t* wo,
                    const float* g, const float* bb,
                    const int* mask_seq, int causal, const Work& W)
{
  const int M = TB * SEQ;                    // 4096
  const long long SB = (long long)SEQ * DM;  // per-batch act block
  const long long SA = (long long)NH * SEQ * SEQ; // per-b attn block
  const long long SH = (long long)SEQ * SEQ; // per-head attn block

  launch_gemm(st, x,    wq, W.QH, nullptr, nullptr, M, DM, DM, DM, DM, DM,
              1, 1, 0, 0, 0, 0, 0, 0, 1.f, B_PKN, false);
  launch_gemm(st, kvin, wk, W.KH, nullptr, nullptr, M, DM, DM, DM, DM, DM,
              1, 1, 0, 0, 0, 0, 0, 0, 1.f, B_PKN, false);
  launch_gemm(st, kvin, wv, W.VH, nullptr, nullptr, M, DM, DM, DM, DM, DM,
              1, 1, 0, 0, 0, 0, 0, 0, 1.f, B_PKN, false);
  // scores: per (b,h)  Q[512,64] @ K^T -> ATT[b,h,512,512], alpha = 1/8
  launch_gemm(st, W.QH, W.KH, W.ATT, nullptr, nullptr, SEQ, SEQ, HD,
              DM, DM, SEQ, TB * NH, NH,
              SB, HD, SB, HD, SA, SH, 0.125f, B_F32T, false);
  softmax_mask_kernel<<<(TB * NH * SEQ) / 8, 256, 0, st>>>(W.ATT, mask_seq,
                                                           causal);
  // context: per (b,h)  P[512,512] @ V[512,64] -> AO cols h*64..  (BNT=64)
  launch_gemm(st, W.ATT, W.VH, W.AO, nullptr, nullptr, SEQ, HD, SEQ,
              SEQ, DM, DM, TB * NH, NH,
              SA, SH, SB, HD, SB, HD, 1.f, B_F32N, false);
  // output proj + residual, then LN in place
  launch_gemm(st, W.AO, wo, x, nullptr, x, M, DM, DM, DM, DM, DM,
              1, 1, 0, 0, 0, 0, 0, 0, 1.f, B_PKN, false);
  layernorm_kernel<<<M / 8, 256, 0, st>>>(x, g, bb);
}

static void run_ffn(hipStream_t st, float* x, const uint32_t* w1,
                    const float* b1, const uint32_t* w2, const float* b2,
                    const float* g, const float* bb, const Work& W)
{
  const int M = TB * SEQ;
  launch_gemm(st, x, w1, W.HID, b1, nullptr, M, DFF, DM, DM, DFF, DFF,
              1, 1, 0, 0, 0, 0, 0, 0, 1.f, B_PKN, true /*relu*/);
  launch_gemm(st, W.HID, w2, x, b2, x, M, DM, DFF, DFF, DM, DM,
              1, 1, 0, 0, 0, 0, 0, 0, 1.f, B_PKN, false);
  layernorm_kernel<<<M / 8, 256, 0, st>>>(x, g, bb);
}

static void pack_rows(hipStream_t st, uint32_t* out, const float* in,
                      long long rows, int N)
{
  const long long pairs = (rows / 2) * (long long)N;
  pack_rows_kernel<<<(unsigned)((pairs + 255) / 256), 256, 0, st>>>(
      out, in, N, pairs);
}

extern "C" void kernel_launch(void* const* d_in, const int* in_sizes, int n_in,
                              void* d_out, int out_size, void* d_ws,
                              size_t ws_size, hipStream_t stream)
{
  const int*   src = (const int*)d_in[IN_SRC];
  const int*   trg = (const int*)d_in[IN_TRG];
  const float* emb = (const float*)d_in[IN_EMB];

  // workspace carve-up: 151 MB f32 activations + 121 MB packed weights
  float* ws = (float*)d_ws;
  const size_t ACT = (size_t)TB * SEQ * DM;        // 2,097,152
  Work W;
  W.QH  = ws;               W.KH = W.QH + ACT;     W.VH = W.KH + ACT;
  W.AO  = W.VH + ACT;       W.X  = W.AO + ACT;     W.Y  = W.X + ACT;
  W.ATT = W.Y + ACT;                                // 16,777,216 floats
  W.HID = W.ATT + (size_t)TB * NH * SEQ * SEQ;      //  8,388,608 floats

  uint32_t* pk = (uint32_t*)(W.HID + (size_t)TB * SEQ * DFF);
  const size_t PK_ATT = (size_t)NL * DM * DM / 2;   // 786,432 dwords / tensor
  const size_t PK_FF  = (size_t)NL * DM * DFF / 2;  // 3,145,728 dwords
  for (int t = 0; t < 12; ++t) W.pkAttn[t] = pk + t * PK_ATT;
  W.pkW1e = pk + 12 * PK_ATT;
  W.pkW2e = W.pkW1e + PK_FF;
  W.pkW1d = W.pkW2e + PK_FF;
  W.pkW2d = W.pkW1d + PK_FF;
  W.pkEmb = W.pkW2d + PK_FF;                        // 8,192,000 dwords

  // ---- pre-pack all weights to bf16 K-pair dwords (once per launch) ----
  const int attnIdx[12] = {EA_WQ, EA_WK, EA_WV, EA_WO,
                           DS_WQ, DS_WK, DS_WV, DS_WO,
                           DX_WQ, DX_WK, DX_WV, DX_WO};
  for (int t = 0; t < 12; ++t)
    pack_rows(stream, W.pkAttn[t], (const float*)d_in[attnIdx[t]],
              (long long)NL * DM, DM);
  pack_rows(stream, W.pkW1e, (const float*)d_in[EF_W1], (long long)NL * DM,
            DFF);
  pack_rows(stream, W.pkW2e, (const float*)d_in[EF_W2], (long long)NL * DFF,
            DM);
  pack_rows(stream, W.pkW1d, (const float*)d_in[DF_W1], (long long)NL * DM,
            DFF);
  pack_rows(stream, W.pkW2d, (const float*)d_in[DF_W2], (long long)NL * DFF,
            DM);
  {
    const long long pairs = (long long)NV * DM / 2;   // emb, flat K-pairs
    pack_flat_kernel<<<(unsigned)((pairs + 255) / 256), 256, 0, stream>>>(
        W.pkEmb, emb, pairs);
  }

  const size_t LA = (size_t)DM * DM / 2;    // per-layer packed attn stride
  const size_t LF = (size_t)DM * DFF / 2;   // per-layer packed ffn stride
  const int M = TB * SEQ;

  // ---------------- encoder ----------------
  embed_pos_kernel<<<(unsigned)(ACT / 256), 256, 0, stream>>>(W.X, emb, src);
  layernorm_kernel<<<M / 8, 256, 0, stream>>>(
      W.X, (const float*)d_in[IN_ENC_G], (const float*)d_in[IN_ENC_B]);
  for (int i = 0; i < NL; ++i) {
    run_mha(stream, W.X, W.X,
            W.pkAttn[0] + i * LA, W.pkAttn[1] + i * LA,
            W.pkAttn[2] + i * LA, W.pkAttn[3] + i * LA,
            (const float*)d_in[EA_G] + i * DM,
            (const float*)d_in[EA_B] + i * DM, src, 0, W);
    run_ffn(stream, W.X,
            W.pkW1e + i * LF, (const float*)d_in[EF_B1] + i * DFF,
            W.pkW2e + i * LF, (const float*)d_in[EF_B2] + i * DM,
            (const float*)d_in[EF_G] + i * DM,
            (const float*)d_in[EF_B] + i * DM, W);
  }
  // ---------------- decoder ----------------
  embed_pos_kernel<<<(unsigned)(ACT / 256), 256, 0, stream>>>(W.Y, emb, trg);
  layernorm_kernel<<<M / 8, 256, 0, stream>>>(
      W.Y, (const float*)d_in[IN_DEC_G], (const float*)d_in[IN_DEC_B]);
  for (int i = 0; i < NL; ++i) {
    run_mha(stream, W.Y, W.Y,
            W.pkAttn[4] + i * LA, W.pkAttn[5] + i * LA,
            W.pkAttn[6] + i * LA, W.pkAttn[7] + i * LA,
            (const float*)d_in[DS_G] + i * DM,
            (const float*)d_in[DS_B] + i * DM, trg, 1 /*causal*/, W);
    run_mha(stream, W.Y, W.X /*enc_out*/,
            W.pkAttn[8] + i * LA, W.pkAttn[9] + i * LA,
            W.pkAttn[10] + i * LA, W.pkAttn[11] + i * LA,
            (const float*)d_in[DX_G] + i * DM,
            (const float*)d_in[DX_B] + i * DM, src, 0, W);
    run_ffn(stream, W.Y,
            W.pkW1d + i * LF, (const float*)d_in[DF_B1] + i * DFF,
            W.pkW2d + i * LF, (const float*)d_in[DF_B2] + i * DM,
            (const float*)d_in[DF_G] + i * DM,
            (const float*)d_in[DF_B] + i * DM, W);
  }
  // ---------------- shared-weight projection: (y @ emb^T) * D^-0.5 ----------
  launch_gemm(stream, W.Y, W.pkEmb, (float*)d_out, nullptr, nullptr,
              M, NV, DM, DM, DM / 2 /*dwords*/, NV, 1, 1,
              0, 0, 0, 0, 0, 0,
              0.04419417382415922f /*1/sqrt(512)*/, B_PKT, false);

  (void)in_sizes; (void)n_in; (void)out_size; (void)ws_size;
}
